// CrossAttention_15874199126366
// MI455X (gfx1250) — matrix operand order repaired and checked by reading, not compiled, not run on hardware
//
#include <hip/hip_runtime.h>

#define HEADS 16
#define HD 64          // head_dim
#define DMODEL 1024    // model dim
#define THREADS 256

// GEMM workgroup tile: 128 rows x 256 cols, K-step 32.
// 8 waves arranged 4(M) x 2(N); each wave: 2 M-frags x 8 N-frags of 16x16.
#define BM 128
#define BN 256
#define BK 32

typedef __attribute__((ext_vector_type(16))) __bf16          v16bf;
typedef __attribute__((ext_vector_type(8)))  float           v8f;
typedef __attribute__((ext_vector_type(8)))  unsigned short  u16x8;
typedef __attribute__((ext_vector_type(4)))  unsigned short  u16x4;

// pointer types for the async builtin: pointee is a GCC-vector int4
typedef __attribute__((vector_size(4 * sizeof(int)))) int v4i_t;
typedef __attribute__((address_space(1))) v4i_t* gptr_v4i;
typedef __attribute__((address_space(3))) v4i_t* lptr_v4i;

union FragBF { v16bf bf; u16x8 h[2]; };

__device__ __forceinline__ unsigned short f2bf(float f) {
  unsigned int u = __float_as_uint(f);
  u += 0x7FFFu + ((u >> 16) & 1u);   // round-to-nearest-even
  return (unsigned short)(u >> 16);
}
__device__ __forceinline__ float bf_lo(unsigned int u) { return __uint_as_float(u << 16); }
__device__ __forceinline__ float bf_hi(unsigned int u) { return __uint_as_float(u & 0xFFFF0000u); }

// ---------------- CDNA5 async global->LDS copy (ASYNCcnt) ------------------
#if defined(__has_builtin)
#  if __has_builtin(__builtin_amdgcn_global_load_async_to_lds_b128)
#    define HAVE_ASYNC_LDS 1
#  endif
#endif

#ifdef HAVE_ASYNC_LDS
__device__ __forceinline__ void async_copy_b128(const void* gsrc, void* ldst) {
  __builtin_amdgcn_global_load_async_to_lds_b128(
      (gptr_v4i)(gsrc), (lptr_v4i)(ldst), 0, 0);
}
__device__ __forceinline__ void async_wait_all() {
#  if __has_builtin(__builtin_amdgcn_s_wait_asynccnt)
  __builtin_amdgcn_s_wait_asynccnt(0);
#  else
  asm volatile("s_wait_asynccnt 0x0" ::: "memory");
#  endif
}
#else
__device__ __forceinline__ void async_copy_b128(const void* gsrc, void* ldst) {
  *(u16x8*)ldst = *(const u16x8*)gsrc;   // sync fallback
}
__device__ __forceinline__ void async_wait_all() {}
#endif

// ---------------------------------------------------------------------------
// Kernel 0: f32 -> bf16 conversion (weights once, activations once).
// ---------------------------------------------------------------------------
__global__ void __launch_bounds__(THREADS) cvt_bf16_kernel(
    const float* __restrict__ src, unsigned short* __restrict__ dst) {
  const int i = (blockIdx.x * THREADS + threadIdx.x) * 4;
  const float4 v = *(const float4*)(src + i);
  u16x4 o;
  o[0] = f2bf(v.x); o[1] = f2bf(v.y); o[2] = f2bf(v.z); o[3] = f2bf(v.w);
  *(u16x4*)(dst + i) = o;
}

// ---------------------------------------------------------------------------
// Shared WMMA GEMM core: acc[2][8] += A_tile(BM x K) * W_tile(K x BN), with
// W row-major [out=1024][in=1024] so B[k][n] = W[n][k]. A,W bf16 in global.
// Double-buffered LDS staging via async b128; static buffer offsets.
// ---------------------------------------------------------------------------
__device__ __forceinline__ void gemm_core(const unsigned short* __restrict__ A,
                                          const unsigned short* __restrict__ W,
                                          int mBase, int nBase,
                                          unsigned short* Asmem,   // [2][BM*BK]
                                          unsigned short* Bsmem,   // [2][BN*BK]
                                          v8f acc[2][8]) {
  const int tid  = threadIdx.x;
  const int lane = tid & 31;
  const int wid  = tid >> 5;
  const int half = lane >> 4;
  const int l16  = lane & 15;
  const int wm   = wid & 3;    // M quadrant (32 rows)
  const int wn   = wid >> 2;   // N half (128 cols)

  // staging assignment: A: 2 threads/row (32B each); B: 1 thread/row (64B)
  const int arow = tid >> 1;
  const int aseg = (tid & 1) * 16;   // element offset
  const int brow = tid;

  const unsigned short* gaBase = A + (size_t)(mBase + arow) * DMODEL + aseg;
  const unsigned short* gbBase = W + (size_t)(nBase + brow) * DMODEL;
  unsigned short* laBase = Asmem + arow * BK + aseg;
  unsigned short* lbBase = Bsmem + brow * BK;

  auto issue_stage = [&](int bufOffA, int bufOffB, int kb) {
    const unsigned short* ga = gaBase + kb;
    unsigned short* la = laBase + bufOffA;
    async_copy_b128(ga,     la);
    async_copy_b128(ga + 8, la + 8);
    const unsigned short* gb = gbBase + kb;
    unsigned short* lb = lbBase + bufOffB;
    async_copy_b128(gb,      lb);
    async_copy_b128(gb + 8,  lb + 8);
    async_copy_b128(gb + 16, lb + 16);
    async_copy_b128(gb + 24, lb + 24);
  };

  const unsigned short* ArowP = Asmem + (wm * 32 + l16) * BK;
  const unsigned short* BrowP = Bsmem + (wn * 128 + l16) * BK;

  auto compute = [&](int bufOffA, int bufOffB) {
    const unsigned short* Arow = ArowP + bufOffA;
    FragBF a0, a1;  // CDNA5 16-bit A layout: VGPR0-3 k in [half*8,+8), VGPR4-7 +16
    a0.h[0] = *(const u16x8*)(Arow + half * 8);
    a0.h[1] = *(const u16x8*)(Arow + 16 + half * 8);
    a1.h[0] = *(const u16x8*)(Arow + 16 * BK + half * 8);
    a1.h[1] = *(const u16x8*)(Arow + 16 * BK + 16 + half * 8);
    #pragma unroll
    for (int nt = 0; nt < 8; ++nt) {
      const unsigned short* Brow = BrowP + bufOffB + nt * (16 * BK);
      FragBF b;  // B layout: k in [half*8,+8), then +16 (matches A)
      b.h[0] = *(const u16x8*)(Brow + half * 8);
      b.h[1] = *(const u16x8*)(Brow + 16 + half * 8);
      acc[0][nt] = __builtin_amdgcn_wmma_f32_16x16x32_bf16(
          false, a0.bf, false, b.bf, (short)0, acc[0][nt], false, false);
      acc[1][nt] = __builtin_amdgcn_wmma_f32_16x16x32_bf16(
          false, a1.bf, false, b.bf, (short)0, acc[1][nt], false, false);
    }
  };

  issue_stage(0, 0, 0);
  // K loop unrolled x2 so double-buffer offsets are compile-time constants.
  for (int kb = 0; kb < DMODEL; kb += 2 * BK) {
    async_wait_all();
    __syncthreads();
    if (kb + BK < DMODEL) issue_stage(BM * BK, BN * BK, kb + BK);
    compute(0, 0);

    async_wait_all();
    __syncthreads();
    if (kb + 2 * BK < DMODEL) issue_stage(0, 0, kb + 2 * BK);
    compute(BM * BK, BN * BK);
  }
}

// ---------------------------------------------------------------------------
// Kernel 1: Q/K/V projections.  blockIdx.z selects matrix. Output bf16 QKV ws.
// ---------------------------------------------------------------------------
__global__ void __launch_bounds__(THREADS, 1) proj_kernel(
    const unsigned short* __restrict__ qact,
    const unsigned short* __restrict__ kact,
    const unsigned short* __restrict__ vact,
    const unsigned short* __restrict__ wbf,   // [3][1024][1024]
    const float* __restrict__ bq,
    const float* __restrict__ bk,
    const float* __restrict__ bv,
    unsigned short* __restrict__ qkv) {       // [3][T][1024]
  __shared__ unsigned short Asmem[2 * BM * BK];   // 16 KB
  __shared__ unsigned short Bsmem[2 * BN * BK];   // 32 KB
  const int m3 = blockIdx.z;
  const unsigned short* A = (m3 == 0) ? qact : (m3 == 1) ? kact : vact;
  const float* bias       = (m3 == 0) ? bq   : (m3 == 1) ? bk   : bv;
  const unsigned short* W = wbf + (size_t)m3 * DMODEL * DMODEL;
  const int mBase = blockIdx.x * BM;
  const int nBase = blockIdx.y * BN;
  const int T = gridDim.x * BM;

  v8f acc[2][8] = {};
  gemm_core(A, W, mBase, nBase, Asmem, Bsmem, acc);

  const int tid = threadIdx.x;
  const int lane = tid & 31, wid = tid >> 5, half = lane >> 4, l16 = lane & 15;
  const int wm = wid & 3, wn = wid >> 2;
  unsigned short* outm = qkv + (size_t)m3 * T * DMODEL;
  #pragma unroll
  for (int nt = 0; nt < 8; ++nt) {
    const int col = nBase + wn * 128 + nt * 16 + l16;
    const float bb = bias[col];
    #pragma unroll
    for (int mt = 0; mt < 2; ++mt) {
      #pragma unroll
      for (int r = 0; r < 8; ++r) {
        const int row = mBase + wm * 32 + mt * 16 + half * 8 + r;
        outm[(size_t)row * DMODEL + col] = f2bf(acc[mt][nt][r] + bb);
      }
    }
  }
}

// ---------------------------------------------------------------------------
// Kernel 2: per-token head-mixing attention. One WG = 16 tokens; Q/K/V tile
// (96 KB) async-staged into dynamic LDS; thread = (token, head).
// ---------------------------------------------------------------------------
__global__ void __launch_bounds__(THREADS, 1) attn_kernel(
    const unsigned short* __restrict__ qkv,  // [3][T][1024]
    unsigned short* __restrict__ attn,       // [T][1024]
    int T) {
  extern __shared__ unsigned short smem[];   // [3][16][1024] = 96 KB
  const int tid = threadIdx.x;
  const int tokenBase = blockIdx.x * 16;

  #pragma unroll
  for (int it = 0; it < 24; ++it) {          // 49152 elems / 256 thr / 8 per b128
    const int e   = (tid + it * THREADS) * 8;
    const int m3  = e >> 14;                 // 16*1024 elems per matrix slice
    const int off = e & 16383;
    const unsigned short* g =
        qkv + (size_t)m3 * T * DMODEL + (size_t)tokenBase * DMODEL + off;
    async_copy_b128(g, smem + e);
  }
  async_wait_all();
  __syncthreads();

  const int t = tid >> 4;
  const int i = tid & 15;
  const unsigned short* Q = smem;
  const unsigned short* K = smem + 16 * DMODEL;
  const unsigned short* V = smem + 32 * DMODEL;

  float qv[HD];
  #pragma unroll
  for (int d = 0; d < HD; d += 2) {
    const unsigned int u = *(const unsigned int*)(Q + t * DMODEL + i * HD + d);
    qv[d] = bf_lo(u); qv[d + 1] = bf_hi(u);
  }
  float s[HEADS];
  #pragma unroll
  for (int j = 0; j < HEADS; ++j) {
    float a = 0.f;
    const unsigned short* kr = K + t * DMODEL + j * HD;
    for (int d = 0; d < HD; d += 2) {
      const unsigned int u = *(const unsigned int*)(kr + d);
      a += qv[d] * bf_lo(u);
      a += qv[d + 1] * bf_hi(u);
    }
    s[j] = a * 0.125f;   // 1/sqrt(64)
  }
  float mx = s[0];
  #pragma unroll
  for (int j = 1; j < HEADS; ++j) mx = fmaxf(mx, s[j]);
  float sum = 0.f;
  #pragma unroll
  for (int j = 0; j < HEADS; ++j) { s[j] = __expf(s[j] - mx); sum += s[j]; }
  const float inv = 1.0f / sum;

  float o[HD];
  #pragma unroll
  for (int d = 0; d < HD; ++d) o[d] = 0.f;
  for (int j = 0; j < HEADS; ++j) {
    const float w = s[j] * inv;
    const unsigned short* vr = V + t * DMODEL + j * HD;
    for (int d = 0; d < HD; d += 2) {
      const unsigned int u = *(const unsigned int*)(vr + d);
      o[d]     += w * bf_lo(u);
      o[d + 1] += w * bf_hi(u);
    }
  }
  unsigned short* op = attn + (size_t)(tokenBase + t) * DMODEL + i * HD;
  for (int d = 0; d < HD; d += 2) {
    *(unsigned int*)(op + d) = ((unsigned int)f2bf(o[d + 1]) << 16) | f2bf(o[d]);
  }
}

// ---------------------------------------------------------------------------
// Kernel 3: output projection -> f32 d_out (+ bias).
// ---------------------------------------------------------------------------
__global__ void __launch_bounds__(THREADS, 1) out_proj_kernel(
    const unsigned short* __restrict__ attn,  // [T][1024]
    const unsigned short* __restrict__ fo_bf, // [1024][1024]
    const float* __restrict__ fo_b,
    float* __restrict__ out) {
  __shared__ unsigned short Asmem[2 * BM * BK];
  __shared__ unsigned short Bsmem[2 * BN * BK];
  const int mBase = blockIdx.x * BM;
  const int nBase = blockIdx.y * BN;

  v8f acc[2][8] = {};
  gemm_core(attn, fo_bf, mBase, nBase, Asmem, Bsmem, acc);

  const int tid = threadIdx.x;
  const int lane = tid & 31, wid = tid >> 5, half = lane >> 4, l16 = lane & 15;
  const int wm = wid & 3, wn = wid >> 2;
  #pragma unroll
  for (int nt = 0; nt < 8; ++nt) {
    const int col = nBase + wn * 128 + nt * 16 + l16;
    const float bb = fo_b[col];
    #pragma unroll
    for (int mt = 0; mt < 2; ++mt) {
      #pragma unroll
      for (int r = 0; r < 8; ++r) {
        const int row = mBase + wm * 32 + mt * 16 + half * 8 + r;
        out[(size_t)row * DMODEL + col] = acc[mt][nt][r] + bb;
      }
    }
  }
}

// ---------------------------------------------------------------------------
extern "C" void kernel_launch(void* const* d_in, const int* in_sizes, int n_in,
                              void* d_out, int out_size, void* d_ws, size_t ws_size,
                              hipStream_t stream) {
  const float* query = (const float*)d_in[0];
  const float* key_  = (const float*)d_in[1];
  const float* value = (const float*)d_in[2];
  const float* wq_w  = (const float*)d_in[3];
  const float* wq_b  = (const float*)d_in[4];
  const float* wk_w  = (const float*)d_in[5];
  const float* wk_b  = (const float*)d_in[6];
  const float* wv_w  = (const float*)d_in[7];
  const float* wv_b  = (const float*)d_in[8];
  const float* fo_w  = (const float*)d_in[9];
  const float* fo_b  = (const float*)d_in[10];

  const int T = in_sizes[0] / DMODEL;          // B*L = 16384
  const size_t M1 = (size_t)DMODEL * DMODEL;   // 1M
  const size_t AC = (size_t)T * DMODEL;        // 16M

  // ws layout (ushort units): weights bf16 | activations bf16 | QKV bf16 | attn bf16
  unsigned short* ws     = (unsigned short*)d_ws;
  unsigned short* w_bf   = ws;                       // 3*M1 (wq,wk,wv)
  unsigned short* fo_bf  = ws + 3 * M1;              // M1
  unsigned short* q_act  = ws + 4 * M1;              // AC
  unsigned short* k_act  = q_act + AC;               // AC
  unsigned short* v_act  = k_act + AC;               // AC
  unsigned short* qkv    = v_act + AC;               // 3*AC
  unsigned short* attn   = qkv + 3 * AC;             // AC

  // 0) conversions (f32 -> bf16)
  cvt_bf16_kernel<<<M1 / (4 * THREADS), THREADS, 0, stream>>>(wq_w, w_bf);
  cvt_bf16_kernel<<<M1 / (4 * THREADS), THREADS, 0, stream>>>(wk_w, w_bf + M1);
  cvt_bf16_kernel<<<M1 / (4 * THREADS), THREADS, 0, stream>>>(wv_w, w_bf + 2 * M1);
  cvt_bf16_kernel<<<M1 / (4 * THREADS), THREADS, 0, stream>>>(fo_w, fo_bf);
  cvt_bf16_kernel<<<AC / (4 * THREADS), THREADS, 0, stream>>>(query, q_act);
  cvt_bf16_kernel<<<AC / (4 * THREADS), THREADS, 0, stream>>>(key_, k_act);
  cvt_bf16_kernel<<<AC / (4 * THREADS), THREADS, 0, stream>>>(value, v_act);

  // 1) Q/K/V projections (WMMA)
  proj_kernel<<<dim3(T / BM, DMODEL / BN, 3), THREADS, 0, stream>>>(
      q_act, k_act, v_act, w_bf, wq_b, wk_b, wv_b, qkv);

  // 2) head-mixing attention
  attn_kernel<<<T / 16, THREADS, 3 * 16 * DMODEL * sizeof(unsigned short), stream>>>(
      qkv, attn, T);

  // 3) output projection (WMMA) + bias -> f32
  out_proj_kernel<<<dim3(T / BM, DMODEL / BN), THREADS, 0, stream>>>(
      attn, fo_bf, fo_b, (float*)d_out);
}
